// WeaklySupervisedSegmentationModel_5927054868515
// MI455X (gfx1250) — matrix operand order repaired and checked by reading, not compiled, not run on hardware
//
#include <hip/hip_runtime.h>
#include <math.h>
#include <stdint.h>

typedef _Float16 half_t;
typedef __attribute__((ext_vector_type(16))) _Float16 v16h;
typedef __attribute__((ext_vector_type(8)))  _Float16 v8h;
typedef __attribute__((ext_vector_type(8)))  float    v8f;

#define NB 4
#define NC 21
#define FD 64
#define HWF 224

// ---------------- epilogue flags ----------------
#define EF_RELU  1
#define EF_BN    2
#define EF_RES   4
#define EF_W32   8
#define EF_W16   16
#define EF_ASYNC 32   // A-tile staged with global_load_async_to_lds_b128

struct ConvArgs {
  const half_t* __restrict__ x;     // f16 NCHW activations
  const half_t* __restrict__ w;     // f16 [Cout][Cin*KH*KW]
  const half_t* __restrict__ res;   // f16 residual (same layout as out)
  const float *bn_g, *bn_b, *bn_m, *bn_v;
  const float* bias;
  half_t* out16; float* out32;
  int Cin, H, W, Cout, KH, KW, stride, pad, Ho, Wo, N, K, flags;
};

// Implicit-GEMM conv: C[Cout][B*Ho*Wo] = W[Cout][K] * im2col[K][N]
// 256 threads = 8 waves (4M x 2N); block tile 64x64; wave tile 16x32 (2 WMMA/K-step).
// Weights double-buffered in LDS via async-to-LDS DMA (ASYNCcnt-tracked).
// Fragments loaded as contiguous 16B LDS vectors (B tile stored n-major).
__global__ __launch_bounds__(256) void conv_wmma(ConvArgs a) {
  __shared__ __align__(16) half_t As[2][64][40];   // [buf][m][k], 80B rows
  __shared__ __align__(16) half_t Bs[2][64][40];   // [buf][n][k] (transposed!)
  const int tid  = threadIdx.x;
  const int lane = tid & 31;
  const int wave = tid >> 5;
  const int wm = wave >> 1, wn = wave & 1;        // 4x2 wave grid
  const int n0 = blockIdx.x * 64, m0 = blockIdx.y * 64;
  const int KHW = a.KH * a.KW, HoWo = a.Ho * a.Wo;
  const int lr = lane & 15, hs = lane >> 4;
  const int ar = tid >> 2, ac = (tid & 3) * 8;    // async A: 256 lanes x 16B
  v8f acc0 = {}, acc1 = {};

  auto stageA = [&](int k0, int buf) {
    if (a.flags & EF_ASYNC) {
      // K%32==0 and Cout%64==0 guaranteed by host: no guards, 16B aligned.
      unsigned dst = (unsigned)(uintptr_t)&As[buf][ar][ac];
      const void* src = a.w + (long)(m0 + ar) * a.K + k0 + ac;
      asm volatile("global_load_async_to_lds_b128 %0, %1, off"
                   :: "v"(dst), "v"(src) : "memory");
    } else {
      for (int i = tid; i < 64 * 32; i += 256) {
        int r = i >> 5, c = i & 31;
        int m = m0 + r, k = k0 + c;
        half_t v = (half_t)0.0f;
        if (m < a.Cout && k < a.K) v = a.w[(long)m * a.K + k];
        As[buf][r][c] = v;
      }
    }
  };
  auto stageB = [&](int k0, int buf) {
    for (int i = tid; i < 64 * 32; i += 256) {
      int nn = i >> 5, kk = i & 31;                // n-major store
      int k = k0 + kk, n = n0 + nn;
      half_t v = (half_t)0.0f;
      if (k < a.K && n < a.N) {
        int ci = k / KHW;  int r1 = k - ci * KHW;
        int kh = r1 / a.KW; int kw = r1 - kh * a.KW;
        int b  = n / HoWo;  int r2 = n - b * HoWo;
        int ho = r2 / a.Wo; int wo = r2 - ho * a.Wo;
        int hi = ho * a.stride - a.pad + kh;
        int wi = wo * a.stride - a.pad + kw;
        if ((unsigned)hi < (unsigned)a.H && (unsigned)wi < (unsigned)a.W)
          v = a.x[(((long)b * a.Cin + ci) * a.H + hi) * a.W + wi];
      }
      Bs[buf][nn][kk] = v;
    }
  };

  const int T = (a.K + 31) >> 5;
  stageA(0, 0);
  stageB(0, 0);
  const int arow = wm * 16 + lr;
  const int bcol0 = wn * 32 + lr, bcol1 = bcol0 + 16;
  for (int t = 0; t < T; ++t) {
    const int cur = t & 1, nxt = cur ^ 1;
    // wave's own async LDS writes complete, then make visible to all waves
    asm volatile("s_wait_asynccnt 0x0" ::: "memory");
    __syncthreads();
    if (t + 1 < T) {                 // software pipeline: prefetch next K-slab
      stageA((t + 1) << 5, nxt);
      stageB((t + 1) << 5, nxt);
    }
    // ---- vectorized fragment loads (ds_load_b128 pairs) ----
    // A lane layout: e=0..15 -> K=16*hs+e
    v8h alo = *(const v8h*)&As[cur][arow][16 * hs];
    v8h ahi = *(const v8h*)&As[cur][arow][16 * hs + 8];
    v16h af = __builtin_shufflevector(alo, ahi, 0, 1, 2, 3, 4, 5, 6, 7,
                                      8, 9, 10, 11, 12, 13, 14, 15);
    // B lane layout (n-major tile): e=0..15 -> K=16*hs+e at fixed column
    v8h b0lo = *(const v8h*)&Bs[cur][bcol0][16 * hs];
    v8h b0hi = *(const v8h*)&Bs[cur][bcol0][16 * hs + 8];
    v16h bf0 = __builtin_shufflevector(b0lo, b0hi, 0, 1, 2, 3, 4, 5, 6, 7,
                                       8, 9, 10, 11, 12, 13, 14, 15);
    v8h b1lo = *(const v8h*)&Bs[cur][bcol1][16 * hs];
    v8h b1hi = *(const v8h*)&Bs[cur][bcol1][16 * hs + 8];
    v16h bf1 = __builtin_shufflevector(b1lo, b1hi, 0, 1, 2, 3, 4, 5, 6, 7,
                                       8, 9, 10, 11, 12, 13, 14, 15);
    acc0 = __builtin_amdgcn_wmma_f32_16x16x32_f16(false, af, false, bf0,
                                                  (short)0, acc0, false, false);
    acc1 = __builtin_amdgcn_wmma_f32_16x16x32_f16(false, af, false, bf1,
                                                  (short)0, acc1, false, false);
  }

  // ---- fused epilogue: BN/bias + residual + ReLU + f16/f32 stores ----
#pragma unroll
  for (int ph = 0; ph < 2; ++ph) {
    v8f& accr = ph ? acc1 : acc0;
#pragma unroll
    for (int vv = 0; vv < 8; ++vv) {
      int m = m0 + wm * 16 + vv + 8 * hs;          // C/D: VGPR v -> M=v(+8)
      int n = n0 + wn * 32 + ph * 16 + lr;         // lane%16 -> N
      if (m >= a.Cout || n >= a.N) continue;
      float val = accr[vv];
      if (a.flags & EF_BN) {
        float s = a.bn_g[m] * rsqrtf(a.bn_v[m] + 1e-5f);
        val = val * s + (a.bn_b[m] - a.bn_m[m] * s);
      } else if (a.bias) {
        val += a.bias[m];
      }
      int b = n / HoWo; int r2 = n - b * HoWo;
      long off = ((long)b * a.Cout + m) * HoWo + r2;
      if (a.flags & EF_RES) val += (float)a.res[off];
      if (a.flags & EF_RELU) val = fmaxf(val, 0.0f);
      if (a.flags & EF_W16) a.out16[off] = (half_t)val;
      if (a.flags & EF_W32) a.out32[off] = val;
    }
  }
}

// ---------------- support kernels ----------------
__global__ void cast_f32_f16(const float* __restrict__ s, half_t* __restrict__ d, long n) {
  for (long i = (long)blockIdx.x * blockDim.x + threadIdx.x; i < n;
       i += (long)gridDim.x * blockDim.x)
    d[i] = (half_t)s[i];
}

__global__ void maxpool3x3s2(const half_t* __restrict__ in, half_t* __restrict__ out,
                             int C, int H, int W, int Ho, int Wo) {
  long total = (long)NB * C * Ho * Wo;
  for (long i = (long)blockIdx.x * blockDim.x + threadIdx.x; i < total;
       i += (long)gridDim.x * blockDim.x) {
    int wo = i % Wo; long t = i / Wo;
    int ho = t % Ho; t /= Ho;
    int c  = t % C;  int b = t / C;
    float m = -3.0e38f;
    for (int kh = 0; kh < 3; ++kh)
      for (int kw = 0; kw < 3; ++kw) {
        int hi = ho * 2 - 1 + kh, wi = wo * 2 - 1 + kw;
        if ((unsigned)hi < (unsigned)H && (unsigned)wi < (unsigned)W)
          m = fmaxf(m, (float)in[(((long)b * C + c) * H + hi) * W + wi]);
      }
    out[i] = (half_t)m;
  }
}

__global__ void gap_k(const float* __restrict__ cam, float* __restrict__ out, int HW) {
  int bc = blockIdx.x * blockDim.x + threadIdx.x;
  if (bc < NB * NC) {
    float s = 0.f;
    for (int i = 0; i < HW; ++i) s += cam[(long)bc * HW + i];
    out[bc] = s / (float)HW;
  }
}

__global__ void resize_bl(const float* __restrict__ in, float* __restrict__ out,
                          int BC, int Hi, int Wi, int Ho, int Wo) {
  long total = (long)BC * Ho * Wo;
  float sh = (float)Hi / Ho, sw = (float)Wi / Wo;
  for (long i = (long)blockIdx.x * blockDim.x + threadIdx.x; i < total;
       i += (long)gridDim.x * blockDim.x) {
    int wo = i % Wo; long t = i / Wo;
    int ho = t % Ho; int bc = t / Ho;
    float fy = (ho + 0.5f) * sh - 0.5f;
    float fx = (wo + 0.5f) * sw - 0.5f;
    int y0 = (int)floorf(fy), x0 = (int)floorf(fx);
    float wy = fy - y0, wx = fx - x0;
    int y1 = y0 + 1, x1 = x0 + 1;
    y0 = min(max(y0, 0), Hi - 1); y1 = min(max(y1, 0), Hi - 1);
    x0 = min(max(x0, 0), Wi - 1); x1 = min(max(x1, 0), Wi - 1);
    const float* p = in + (long)bc * Hi * Wi;
    float v = (1 - wy) * ((1 - wx) * p[y0 * Wi + x0] + wx * p[y0 * Wi + x1]) +
              wy       * ((1 - wx) * p[y1 * Wi + x0] + wx * p[y1 * Wi + x1]);
    out[i] = v;
  }
}

__global__ void chan_norm(float* feat, long HW) {  // in-place, per-(b,pixel)
  long total = (long)NB * HW;
  for (long i = (long)blockIdx.x * blockDim.x + threadIdx.x; i < total;
       i += (long)gridDim.x * blockDim.x) {
    int b = i / HW; long p = i % HW;
    float s = 0.f;
    for (int d = 0; d < FD; ++d) {
      float v = feat[((long)b * FD + d) * HW + p];
      s += v * v;
    }
    float inv = 1.f / (sqrtf(s) + 1e-8f);
    for (int d = 0; d < FD; ++d) feat[((long)b * FD + d) * HW + p] *= inv;
  }
}

__global__ void rg_init(const float* __restrict__ seg, float* __restrict__ refined, long n) {
  for (long i = (long)blockIdx.x * blockDim.x + threadIdx.x; i < n;
       i += (long)gridDim.x * blockDim.x)
    refined[i] = (seg[i] > 0.1f) ? 1.0f : 0.0f;
}

__global__ void rg_fgsum(const float* __restrict__ refined, const int* __restrict__ labels,
                         float* fgsum, int* active, int HW) {
  __shared__ float red[256];
  int bc = blockIdx.x;
  const float* r = refined + (long)bc * HW;
  float s = 0.f;
  for (int i = threadIdx.x; i < HW; i += 256) s += (r[i] > 0.5f) ? 1.f : 0.f;
  red[threadIdx.x] = s; __syncthreads();
  for (int st = 128; st > 0; st >>= 1) {
    if (threadIdx.x < st) red[threadIdx.x] += red[threadIdx.x + st];
    __syncthreads();
  }
  if (threadIdx.x == 0) {
    fgsum[bc] = red[0];
    active[bc] = (red[0] > 0.f) && (labels[bc] > 0);
  }
}

__global__ void rg_proto(const float* __restrict__ refined, const float* __restrict__ normed,
                         const float* __restrict__ fgsum, float* fgmean, int HW) {
  __shared__ float red[256];
  int idx = blockIdx.x;                  // (b*NC + c)*FD + d
  int d = idx % FD; int bc = idx / FD; int b = bc / NC;
  const float* r  = refined + (long)bc * HW;
  const float* nm = normed + ((long)b * FD + d) * HW;
  float s = 0.f;
  for (int i = threadIdx.x; i < HW; i += 256)
    s += ((r[i] > 0.5f) ? 1.f : 0.f) * nm[i];
  red[threadIdx.x] = s; __syncthreads();
  for (int st = 128; st > 0; st >>= 1) {
    if (threadIdx.x < st) red[threadIdx.x] += red[threadIdx.x + st];
    __syncthreads();
  }
  if (threadIdx.x == 0) fgmean[(long)bc * FD + d] = red[0] / (fgsum[bc] + 1e-8f);
}

__global__ void rg_sim(const float* __restrict__ normed, const float* __restrict__ fgmean,
                       float* __restrict__ sim, int HW) {
  __shared__ float fm[NC * FD];
  int b = blockIdx.y;
  for (int i = threadIdx.x; i < NC * FD; i += blockDim.x)
    fm[i] = fgmean[(long)b * NC * FD + i];
  __syncthreads();
  int p = blockIdx.x * blockDim.x + threadIdx.x;
  if (p >= HW) return;
  float accv[NC];
#pragma unroll
  for (int c = 0; c < NC; ++c) accv[c] = 0.f;
  for (int d = 0; d < FD; ++d) {
    float nv = normed[((long)b * FD + d) * HW + p];
#pragma unroll
    for (int c = 0; c < NC; ++c) accv[c] += nv * fm[c * FD + d];
  }
  for (int c = 0; c < NC; ++c) sim[((long)b * NC + c) * HW + p] = accv[c];
}

__global__ void rg_thr(const float* __restrict__ sim, float* thr, float* smax,
                       int HW, float coef) {
  __shared__ float rs[256], rq[256], rm[256];
  int bc = blockIdx.x;
  const float* s = sim + (long)bc * HW;
  float su = 0.f, sq = 0.f, mx = -3.0e38f;
  for (int i = threadIdx.x; i < HW; i += 256) {
    float v = s[i]; su += v; sq += v * v; mx = fmaxf(mx, v);
  }
  rs[threadIdx.x] = su; rq[threadIdx.x] = sq; rm[threadIdx.x] = mx;
  __syncthreads();
  for (int st = 128; st > 0; st >>= 1) {
    if (threadIdx.x < st) {
      rs[threadIdx.x] += rs[threadIdx.x + st];
      rq[threadIdx.x] += rq[threadIdx.x + st];
      rm[threadIdx.x] = fmaxf(rm[threadIdx.x], rm[threadIdx.x + st]);
    }
    __syncthreads();
  }
  if (threadIdx.x == 0) {
    float n = (float)HW;
    float mean = rs[0] / n;
    float var  = fmaxf((rq[0] - rs[0] * rs[0] / n) / (n - 1.0f), 0.f);
    thr[bc] = mean + coef * sqrtf(var);
    smax[bc] = rm[0];
  }
}

__global__ void rg_update(const float* __restrict__ rin, float* __restrict__ rout,
                          const float* __restrict__ sim, const float* __restrict__ thr,
                          const float* __restrict__ smax, const int* __restrict__ active,
                          int H, int W, int it) {
  long total = (long)NB * NC * H * W;
  int HW = H * W;
  for (long i = (long)blockIdx.x * blockDim.x + threadIdx.x; i < total;
       i += (long)gridDim.x * blockDim.x) {
    int bc = i / HW; int p = i % HW;
    int h = p / W, w = p % W;
    float r = rin[i];
    float fg = (r > 0.5f) ? 1.f : 0.f;
    float sv = sim[i], t = thr[bc];
    bool np = (sv > t) && (fg <= 0.5f) && active[bc];
    if (np && it > 0) {
      bool dil = false;
      for (int dh = -1; dh <= 1; ++dh)
        for (int dw = -1; dw <= 1; ++dw) {
          int hh = h + dh, ww = w + dw;
          if ((unsigned)hh < (unsigned)H && (unsigned)ww < (unsigned)W)
            dil |= (rin[(long)bc * HW + hh * W + ww] > 0.5f);
        }
      np = dil;
    }
    float conf = fminf(fmaxf((sv - t) / (smax[bc] - t + 1e-8f), 0.1f), 0.9f);
    rout[i] = r + (np ? conf : 0.f);
  }
}

__global__ void rg_final(const float* __restrict__ refined, float* __restrict__ out, int HW) {
  long total = (long)NB * HW;
  for (long i = (long)blockIdx.x * blockDim.x + threadIdx.x; i < total;
       i += (long)gridDim.x * blockDim.x) {
    int b = i / HW; long p = i % HW;
    float vals[NC]; int arg = 0; float best = -3.0e38f;
#pragma unroll
    for (int c = 0; c < NC; ++c) {
      float v = fminf(fmaxf(refined[((long)b * NC + c) * HW + p], 0.f), 1.f);
      vals[c] = v;
      if (v > best) { best = v; arg = c; }
    }
    for (int c = 0; c < NC; ++c)
      out[((long)b * NC + c) * HW + p] = (c == arg) ? vals[c] : 0.f;
  }
}

// ---------------- host orchestration ----------------
extern "C" void kernel_launch(void* const* d_in, const int* in_sizes, int n_in,
                              void* d_out, int out_size, void* d_ws, size_t ws_size,
                              hipStream_t stream) {
  const float* x_f32 = (const float*)d_in[0];
  const int* labels = (const int*)d_in[1];

  // --- parameter walk (jax tree-flatten order: sorted dict keys) ---
  int pi = 2;
  auto nextp = [&]() -> const float* { return (const float*)d_in[pi++]; };
  struct BNP { const float *b, *g, *m, *v; };
  auto readBN = [&]() { BNP r; r.b = nextp(); r.g = nextp(); r.m = nextp(); r.v = nextp(); return r; };
  BNP bn1 = readBN();
  const float* w_conv1 = nextp();
  const float* cs_b = nextp(); const float* cs_w = nextp();
  const float* fp_b = nextp(); const float* fp_w = nextp();
  struct Blk { BNP bn1, bn2, bn3, dbn; const float *c1, *c2, *c3, *down; };
  auto readBlk = [&](bool down) {
    Blk k{}; k.bn1 = readBN(); k.bn2 = readBN(); k.bn3 = readBN();
    k.c1 = nextp(); k.c2 = nextp(); k.c3 = nextp();
    if (down) { k.dbn = readBN(); k.down = nextp(); } else k.down = nullptr;
    return k;
  };
  Blk L1[3], L2[4], L3[6], L4[3];
  for (int i = 0; i < 3; ++i) L1[i] = readBlk(i == 0);
  for (int i = 0; i < 4; ++i) L2[i] = readBlk(i == 0);
  for (int i = 0; i < 6; ++i) L3[i] = readBlk(i == 0);
  for (int i = 0; i < 3; ++i) L4[i] = readBlk(i == 0);

  // --- workspace carve ---
  char* base = (char*)d_ws;
  size_t off = 0;
  auto alloc = [&](size_t bytes) {
    size_t p = (off + 255) & ~(size_t)255; off = p + bytes; return (void*)(base + p);
  };
  half_t* warena = (half_t*)alloc((size_t)26 * 1024 * 1024 * 2); // f16 weight arena
  size_t woff = 0;
  auto castw = [&](const float* w, size_t n) {
    half_t* d = warena + woff; woff += (n + 127) & ~(size_t)127;
    long blocks = (long)((n + 255) / 256);
    cast_f32_f16<<<dim3((unsigned)blocks), dim3(256), 0, stream>>>(w, d, (long)n);
    return d;
  };
  const size_t SLOT = 3211264;  // max f16 activation elems (64*112*112*4)
  half_t* slot[6];
  for (int i = 0; i < 6; ++i) slot[i] = (half_t*)alloc(SLOT * sizeof(half_t));
  const int HWp = HWF * HWF;                       // 50176
  float* cam    = (float*)alloc((size_t)NB * NC * 49 * 4);
  float* proj   = (float*)alloc((size_t)NB * FD * 196 * 4);
  float* seg    = (float*)alloc((size_t)NB * NC * HWp * 4);
  float* featrs = (float*)alloc((size_t)NB * FD * HWp * 4);
  float* refA   = (float*)alloc((size_t)NB * NC * HWp * 4);
  float* refB   = (float*)alloc((size_t)NB * NC * HWp * 4);
  float* simb   = (float*)alloc((size_t)NB * NC * HWp * 4);
  float* fgmean = (float*)alloc((size_t)NB * NC * FD * 4);
  float* fgsum  = (float*)alloc((size_t)NB * NC * 4);
  float* thr    = (float*)alloc((size_t)NB * NC * 4);
  float* smax   = (float*)alloc((size_t)NB * NC * 4);
  int*   activ  = (int*)alloc((size_t)NB * NC * 4);

  // --- conv launcher ---
  auto conv = [&](const half_t* xin, int Cin, int H, int W, const half_t* w,
                  int Cout, int KH, int KW, int st, int pad, const BNP* bn,
                  const float* bias, const half_t* res, int flags,
                  half_t* o16, float* o32, int& Ho, int& Wo) {
    Ho = (H + 2 * pad - KH) / st + 1;
    Wo = (W + 2 * pad - KW) / st + 1;
    ConvArgs a{};
    a.x = xin; a.w = w; a.res = res;
    if (bn) { a.bn_g = bn->g; a.bn_b = bn->b; a.bn_m = bn->m; a.bn_v = bn->v; }
    a.bias = bias; a.out16 = o16; a.out32 = o32;
    a.Cin = Cin; a.H = H; a.W = W; a.Cout = Cout; a.KH = KH; a.KW = KW;
    a.stride = st; a.pad = pad; a.Ho = Ho; a.Wo = Wo;
    a.N = NB * Ho * Wo; a.K = Cin * KH * KW; a.flags = flags;
    if ((a.K % 32 == 0) && (Cout % 64 == 0)) a.flags |= EF_ASYNC;
    dim3 grid((a.N + 63) / 64, (Cout + 63) / 64);
    conv_wmma<<<grid, dim3(256), 0, stream>>>(a);
  };

  // --- cast all weights to f16 ---
  half_t* h_conv1 = castw(w_conv1, (size_t)64 * 3 * 49);
  half_t* h_cs = castw(cs_w, (size_t)NC * 2048);
  half_t* h_fp = castw(fp_w, (size_t)FD * 1024);
  struct BlkH { half_t *c1, *c2, *c3, *down; };
  BlkH H1[3], H2[4], H3[6], H4[3];
  auto castBlk = [&](const Blk& b, int inC, int planes) {
    BlkH h{};
    h.c1 = castw(b.c1, (size_t)planes * inC);
    h.c2 = castw(b.c2, (size_t)planes * planes * 9);
    h.c3 = castw(b.c3, (size_t)planes * 4 * planes);
    h.down = b.down ? castw(b.down, (size_t)planes * 4 * inC) : nullptr;
    return h;
  };
  { int inC = 64;   for (int i = 0; i < 3; ++i) { H1[i] = castBlk(L1[i], inC, 64);  inC = 256;  } }
  { int inC = 256;  for (int i = 0; i < 4; ++i) { H2[i] = castBlk(L2[i], inC, 128); inC = 512;  } }
  { int inC = 512;  for (int i = 0; i < 6; ++i) { H3[i] = castBlk(L3[i], inC, 256); inC = 1024; } }
  { int inC = 1024; for (int i = 0; i < 3; ++i) { H4[i] = castBlk(L4[i], inC, 512); inC = 2048; } }

  // --- stem ---
  cast_f32_f16<<<dim3(1176), dim3(256), 0, stream>>>(x_f32, slot[5], (long)NB * 3 * HWp);
  int Ho, Wo;
  conv(slot[5], 3, 224, 224, h_conv1, 64, 7, 7, 2, 3, &bn1, nullptr, nullptr,
       EF_BN | EF_RELU | EF_W16, slot[0], nullptr, Ho, Wo);           // 64x112x112
  maxpool3x3s2<<<dim3(3136), dim3(256), 0, stream>>>(slot[0], slot[1], 64, 112, 112, 56, 56);

  half_t* cur = slot[1];
  half_t* pp  = slot[0];
  half_t* t1 = slot[2]; half_t* t2 = slot[3]; half_t* idn = slot[4];
  int C = 64, H = 56, W = 56;

  auto bottleneck = [&](const Blk& bk, const BlkH& hk, int planes, int stride) {
    int h2, w2, ho, wo;
    conv(cur, C, H, W, hk.c1, planes, 1, 1, 1, 0, &bk.bn1, nullptr, nullptr,
         EF_BN | EF_RELU | EF_W16, t1, nullptr, h2, w2);
    conv(t1, planes, H, W, hk.c2, planes, 3, 3, stride, 1, &bk.bn2, nullptr, nullptr,
         EF_BN | EF_RELU | EF_W16, t2, nullptr, ho, wo);
    const half_t* resid = cur;
    if (hk.down) {
      int a1, a2;
      conv(cur, C, H, W, hk.down, planes * 4, 1, 1, stride, 0, &bk.dbn, nullptr, nullptr,
           EF_BN | EF_W16, idn, nullptr, a1, a2);
      resid = idn;
    }
    conv(t2, planes, ho, wo, hk.c3, planes * 4, 1, 1, 1, 0, &bk.bn3, nullptr, resid,
         EF_BN | EF_RES | EF_RELU | EF_W16, pp, nullptr, ho, wo);
    half_t* tmp = cur; cur = pp; pp = tmp;
    C = planes * 4; H = ho; W = wo;
  };

  for (int i = 0; i < 3; ++i) bottleneck(L1[i], H1[i], 64, 1);
  for (int i = 0; i < 4; ++i) bottleneck(L2[i], H2[i], 128, i == 0 ? 2 : 1);
  for (int i = 0; i < 6; ++i) bottleneck(L3[i], H3[i], 256, i == 0 ? 2 : 1);
  // x3 = cur (1024x14x14): project now before layer4 recycles buffers
  conv(cur, 1024, 14, 14, h_fp, FD, 1, 1, 1, 0, nullptr, fp_b, nullptr,
       EF_W32, nullptr, proj, Ho, Wo);
  for (int i = 0; i < 3; ++i) bottleneck(L4[i], H4[i], 512, i == 0 ? 2 : 1);
  // cam head on x4 (2048x7x7)
  conv(cur, 2048, 7, 7, h_cs, NC, 1, 1, 1, 0, nullptr, cs_b, nullptr,
       EF_W32 | EF_RELU, nullptr, cam, Ho, Wo);

  float* logits = (float*)d_out;
  float* segout = (float*)d_out + NB * NC;
  gap_k<<<dim3(1), dim3(128), 0, stream>>>(cam, logits, 49);

  resize_bl<<<dim3(16464), dim3(256), 0, stream>>>(cam, seg, NB * NC, 7, 7, HWF, HWF);
  resize_bl<<<dim3(16464), dim3(256), 0, stream>>>(proj, featrs, NB * FD, 14, 14, HWF, HWF);
  chan_norm<<<dim3(784), dim3(256), 0, stream>>>(featrs, (long)HWp);
  rg_init<<<dim3(16464), dim3(256), 0, stream>>>(seg, refA, (long)NB * NC * HWp);

  for (int it = 0; it < 5; ++it) {
    float* rin  = (it & 1) ? refB : refA;
    float* rout = (it & 1) ? refA : refB;
    rg_fgsum<<<dim3(NB * NC), dim3(256), 0, stream>>>(rin, labels, fgsum, activ, HWp);
    rg_proto<<<dim3(NB * NC * FD), dim3(256), 0, stream>>>(rin, featrs, fgsum, fgmean, HWp);
    rg_sim<<<dim3((HWp + 255) / 256, NB), dim3(256), 0, stream>>>(featrs, fgmean, simb, HWp);
    rg_thr<<<dim3(NB * NC), dim3(256), 0, stream>>>(simb, thr, smax, HWp, 0.5f - 0.05f * it);
    rg_update<<<dim3(16464), dim3(256), 0, stream>>>(rin, rout, simb, thr, smax, activ,
                                                     HWF, HWF, it);
  }
  // after 5 iters final result lives in refB
  rg_final<<<dim3(784), dim3(256), 0, stream>>>(refB, segout, HWp);
}
